// LawinAttn_13099650253378
// MI455X (gfx1250) — matrix operand order repaired and checked by reading, not compiled, not run on hardware
//
#include <hip/hip_runtime.h>
#include <hip/hip_bf16.h>

typedef _Float16 h16;
typedef __attribute__((ext_vector_type(2)))  _Float16 v2h;
typedef __attribute__((ext_vector_type(4)))  _Float16 v4h;
typedef __attribute__((ext_vector_type(8)))  _Float16 v8h;
typedef __attribute__((ext_vector_type(16))) _Float16 v16h;
typedef __attribute__((ext_vector_type(8)))  float    v8f;
typedef __attribute__((ext_vector_type(2)))  float    v2f;
typedef __attribute__((ext_vector_type(4)))  float    v4f;

// ---------------- geometry ----------------
#define NBATCH 1024
#define C512   512
#define IC256  256
#define P64    64
#define HEADS  64

// LDS layout (bytes). One workgroup per batch. f16 intermediates.
#define CT_STR 520           // padded row stride (f16) for transposed [p][c] buffers
#define TP_STR 264           // padded row stride for thT/phT/yT [p][ic]
#define AT_STR 72            // padded row stride for per-wave attn [q][k]
#define OFF_CTXT   0                       // 64*520*2  = 66560
#define OFF_QT     66560                   // 64*520*2  = 66560
#define OFF_THT    133120                  // 64*264*2  = 33792 (theta, transposed [p][ic])
#define OFF_PHT    166912                  // 33792     (phi, transposed [p][ic])
#define OFF_GX     200704                  // 256*64*2  = 32768 (g, row-major [ic][p])
#define OFF_CTXROW OFF_THT                 // 512*64*2 = 65536 (aliases THT+PHT, dead before P3)
#define OFF_ATTN   0                       // 8 waves * 64*72*2 = 73728 (aliases CTXT, dead after P3)
#define OFF_YT     73728                   // 64*264*2 = 33792 (inside QT region, dead after P3)
#define SMEM_BYTES 233472

// ws layout (bytes)
#define WOFF_PMW  0u          // 64*64*64 f16
#define WOFF_GW   524288u     // 256*512 f16
#define WOFF_THW  786432u
#define WOFF_PHW  1048576u
#define WOFF_OUTW 1310720u    // 512*256 f16
#define WOFF_GSC  1572864u
#define WOFF_GSH  1573888u
#define WOFF_TSC  1574912u
#define WOFF_TSH  1575936u
#define WOFF_PSC  1576960u
#define WOFF_PSH  1577984u
#define WOFF_OSC  1579008u
#define WOFF_OSH  1581056u

static __device__ __forceinline__ v8f zero8f() {
    v8f z;
#pragma unroll
    for (int i = 0; i < 8; ++i) z[i] = 0.0f;
    return z;
}
static __device__ __forceinline__ v16h cat8(v8h lo, v8h hi) {
    return __builtin_shufflevector(lo, hi, 0,1,2,3,4,5,6,7,8,9,10,11,12,13,14,15);
}
static __device__ __forceinline__ v8f wmma16(v16h a, v16h b, v8f c) {
    return __builtin_amdgcn_wmma_f32_16x16x32_f16(false, a, false, b, (short)0, c, false, false);
}

// ---------------- prep: fp32 weights -> f16, fold BN into scale/shift ----------------
__global__ __launch_bounds__(256) void lawin_prep(
    const float* __restrict__ pm_W,
    const float* __restrict__ g_W,  const float* __restrict__ th_W,
    const float* __restrict__ ph_W, const float* __restrict__ out_W,
    const float* __restrict__ g_gamma, const float* __restrict__ g_beta,
    const float* __restrict__ g_mean,  const float* __restrict__ g_var,
    const float* __restrict__ t_gamma, const float* __restrict__ t_beta,
    const float* __restrict__ t_mean,  const float* __restrict__ t_var,
    const float* __restrict__ p_gamma, const float* __restrict__ p_beta,
    const float* __restrict__ p_mean,  const float* __restrict__ p_var,
    const float* __restrict__ o_gamma, const float* __restrict__ o_beta,
    const float* __restrict__ o_mean,  const float* __restrict__ o_var,
    char* __restrict__ ws)
{
    h16* pmW16  = (h16*)(ws + WOFF_PMW);
    h16* gW16   = (h16*)(ws + WOFF_GW);
    h16* thW16  = (h16*)(ws + WOFF_THW);
    h16* phW16  = (h16*)(ws + WOFF_PHW);
    h16* outW16 = (h16*)(ws + WOFF_OUTW);
    float* gSc = (float*)(ws + WOFF_GSC); float* gSh = (float*)(ws + WOFF_GSH);
    float* tSc = (float*)(ws + WOFF_TSC); float* tSh = (float*)(ws + WOFF_TSH);
    float* pSc = (float*)(ws + WOFF_PSC); float* pSh = (float*)(ws + WOFF_PSH);
    float* oSc = (float*)(ws + WOFF_OSC); float* oSh = (float*)(ws + WOFF_OSH);

    int gid = blockIdx.x * 256 + threadIdx.x;
    int gsz = gridDim.x * 256;
    for (int i = gid; i < HEADS * P64 * P64; i += gsz) pmW16[i] = (h16)pm_W[i];
    for (int i = gid; i < IC256 * C512; i += gsz) {
        gW16[i]   = (h16)g_W[i];
        thW16[i]  = (h16)th_W[i];
        phW16[i]  = (h16)ph_W[i];
        outW16[i] = (h16)out_W[i];   // out_W is 512*256 == 256*512 elements
    }
    for (int i = gid; i < IC256; i += gsz) {
        float s;
        s = g_gamma[i] * rsqrtf(g_var[i] + 1e-5f); gSc[i] = s; gSh[i] = g_beta[i] - g_mean[i] * s;
        s = t_gamma[i] * rsqrtf(t_var[i] + 1e-5f); tSc[i] = s; tSh[i] = t_beta[i] - t_mean[i] * s;
        s = p_gamma[i] * rsqrtf(p_var[i] + 1e-5f); pSc[i] = s; pSh[i] = p_beta[i] - p_mean[i] * s;
    }
    for (int i = gid; i < C512; i += gsz) {
        float s = o_gamma[i] * rsqrtf(o_var[i] + 1e-5f);
        oSc[i] = s; oSh[i] = o_beta[i] - o_mean[i] * s;
    }
}

// ---------------- fused attention kernel: one workgroup per batch ----------------
__global__ __launch_bounds__(256) void lawin_main(
    const float* __restrict__ query, const float* __restrict__ context,
    const float* __restrict__ pm_b,  const char* __restrict__ ws,
    float* __restrict__ out)
{
    extern __shared__ __align__(16) char smem[];
    h16* sCtxT   = (h16*)(smem + OFF_CTXT);   // ctx2 transposed [p][c], stride CT_STR
    h16* sQT     = (h16*)(smem + OFF_QT);     // query transposed [p][c]
    h16* sThT    = (h16*)(smem + OFF_THT);    // theta proj transposed [p][ic], stride TP_STR
    h16* sPhT    = (h16*)(smem + OFF_PHT);    // phi   proj transposed [p][ic]
    h16* sGx     = (h16*)(smem + OFF_GX);     // g     proj row-major [ic][p]
    h16* sCtxRow = (h16*)(smem + OFF_CTXROW); // staging ctx [c][p]
    h16* sYT     = (h16*)(smem + OFF_YT);     // yT [p][ic], stride TP_STR

    const h16* pmW  = (const h16*)(ws + WOFF_PMW);
    const h16* gW   = (const h16*)(ws + WOFF_GW);
    const h16* thW  = (const h16*)(ws + WOFF_THW);
    const h16* phW  = (const h16*)(ws + WOFF_PHW);
    const h16* outW = (const h16*)(ws + WOFF_OUTW);
    const float* gSc = (const float*)(ws + WOFF_GSC); const float* gSh = (const float*)(ws + WOFF_GSH);
    const float* tSc = (const float*)(ws + WOFF_TSC); const float* tSh = (const float*)(ws + WOFF_TSH);
    const float* pSc = (const float*)(ws + WOFF_PSC); const float* pSh = (const float*)(ws + WOFF_PSH);
    const float* oSc = (const float*)(ws + WOFF_OSC); const float* oSh = (const float*)(ws + WOFF_OSH);

    const int n    = blockIdx.x;
    const int tid  = threadIdx.x;
    const int w    = tid >> 5;
    const int lane = tid & 31;
    const int l15  = lane & 15;
    const int hh   = lane >> 4;               // lane half (wave32)
    const float* qn = query   + (size_t)n * (C512 * P64);
    const float* cn = context + (size_t)n * (C512 * P64);

    // ---- P0: stage query (transposed) and context (row-major) as f16, b128 loads ----
    for (int i = tid * 4; i < C512 * P64; i += 256 * 4) {
        int c = i >> 6, p = i & 63;
        v4f qv = *(const v4f*)(qn + i);
        v4f cv = *(const v4f*)(cn + i);
        v4h cp;
#pragma unroll
        for (int j = 0; j < 4; ++j) {
            sQT[(p + j) * CT_STR + c] = (h16)qv[j];
            cp[j] = (h16)cv[j];
        }
        *(v4h*)(sCtxRow + i) = cp;
    }
    __syncthreads();

    // ---- P1: position mixing, write ctx2 transposed ----
    // per head h: Y[c',q] = sum_p X[c',p] * pmW[h][q][p] ; M=8(pad16), N=64, K=64
    {
        const int base0 = hh ? 16 : 0;
        const int arow = l15 & 7;             // rows 8..15 duplicate 0..7 (D rows 8..15 discarded)
        for (int job = w; job < HEADS * 4; job += 8) {
            int head = job >> 2, qt = job & 3;
            v8f acc = zero8f();
#pragma unroll
            for (int ks = 0; ks < 64; ks += 32) {
                const h16* src = sCtxRow + (head * 8 + arow) * 64 + ks + base0;
                v16h a = cat8(*(const v8h*)src, *(const v8h*)(src + 8));
                int q  = qt * 16 + l15;
                int kb = ks + hh * 16;
                const h16* bs = pmW + head * 4096 + q * 64 + kb;
                v16h b = cat8(*(const v8h*)bs, *(const v8h*)(bs + 8));
                acc = wmma16(a, b, acc);
            }
            if (lane < 16) {                  // D rows 0..7 valid, live in lane half 0
                int q = qt * 16 + lane;
                float bias = pm_b[head * 64 + q];
                v8h res;
#pragma unroll
                for (int r = 0; r < 8; ++r) {
                    float v = acc[r] + bias + (float)sCtxRow[(head * 8 + r) * 64 + q];
                    res[r] = (h16)v;
                }
                *(v8h*)(sCtxT + q * CT_STR + head * 8) = res;   // 16B store
            }
        }
    }
    __syncthreads();

    // ---- P3: g/theta/phi projections (M=256, N=64, K=512), BN+ReLU epilogue ----
    for (int job = w; job < 3 * 64; job += 8) {
        int which = job >> 6;                 // 0:g 1:theta 2:phi
        int t  = job & 63;
        int mt = t >> 2, nt = t & 3;
        const h16*   Wsel = (which == 0) ? gW  : (which == 1) ? thW : phW;
        const h16*   sB   = (which == 1) ? sQT : sCtxT;
        const float* sc   = (which == 0) ? gSc : (which == 1) ? tSc : pSc;
        const float* sh   = (which == 0) ? gSh : (which == 1) ? tSh : pSh;

        int m = mt * 16 + l15;
        int p = nt * 16 + l15;
        v8f acc = zero8f();
#pragma unroll 2
        for (int k0 = 0; k0 < C512; k0 += 32) {
            const h16* as = Wsel + m * C512 + k0 + (hh ? 16 : 0);
            v16h a = cat8(*(const v8h*)as, *(const v8h*)(as + 8));
            const h16* bs = sB + p * CT_STR + k0 + hh * 16;
            v16h b = cat8(*(const v8h*)bs, *(const v8h*)(bs + 8));
            acc = wmma16(a, b, acc);
        }
        if (which == 0) {                     // g: row-major [ic][p]
#pragma unroll
            for (int r = 0; r < 8; ++r) {
                int mr = mt * 16 + r + hh * 8;
                float v = fmaxf(acc[r] * sc[mr] + sh[mr], 0.0f);
                sGx[mr * 64 + p] = (h16)v;
            }
        } else {                              // theta/phi: transposed [p][ic], one b128 store
            h16* sT = (which == 1) ? sThT : sPhT;
            v8h pk;
#pragma unroll
            for (int r = 0; r < 8; ++r) {
                int mr = mt * 16 + r + hh * 8;
                float v = fmaxf(acc[r] * sc[mr] + sh[mr], 0.0f);
                pk[r] = (h16)v;
            }
            *(v8h*)(sT + p * TP_STR + mt * 16 + hh * 8) = pk;
        }
    }
    __syncthreads();

    // ---- P4: per-head attention (each wave owns 8 heads) ----
    {
        h16* sAttn = (h16*)(smem + OFF_ATTN) + w * (64 * AT_STR);
        const int kd = hh * 2;                // f32 WMMA K packing: half0->K{0,1}, half1->K{2,3}
        for (int head = w; head < HEADS; head += 8) {
            int hd = head * 4 + kd;
            // logits^T[k][q] = sum_d ph[d][k] * th[d][q] (f32 WMMA 16x16x4), softmax over k
#pragma unroll 1
            for (int qt = 0; qt < 4; ++qt) {
                int q = qt * 16 + l15;
                v2h tb = *(const v2h*)(sThT + q * TP_STR + hd);   // one b32 load
                v2f bfr; bfr[0] = (float)tb[0]; bfr[1] = (float)tb[1];
                v8f lt[4];
#pragma unroll
                for (int kt = 0; kt < 4; ++kt) {
                    int k = kt * 16 + l15;
                    v2h ta = *(const v2h*)(sPhT + k * TP_STR + hd);
                    v2f afr; afr[0] = (float)ta[0]; afr[1] = (float)ta[1];
                    v8f cz = zero8f();
                    lt[kt] = __builtin_amdgcn_wmma_f32_16x16x4_f32(
                        false, afr, false, bfr, (short)0, cz, false, false);
                }
                // softmax over k: 32 values per lane + partner half via shfl_xor(16)
                float mloc = -3.402823e38f;
#pragma unroll
                for (int kt = 0; kt < 4; ++kt)
#pragma unroll
                    for (int r = 0; r < 8; ++r)
                        mloc = fmaxf(mloc, lt[kt][r] * 0.5f);
                mloc = fmaxf(mloc, __shfl_xor(mloc, 16));
                float ssum = 0.0f;
#pragma unroll
                for (int kt = 0; kt < 4; ++kt)
#pragma unroll
                    for (int r = 0; r < 8; ++r) {
                        float e = __expf(lt[kt][r] * 0.5f - mloc);
                        lt[kt][r] = e;
                        ssum += e;
                    }
                ssum += __shfl_xor(ssum, 16);
                float inv = 1.0f / ssum;
#pragma unroll
                for (int kt = 0; kt < 4; ++kt) {
                    v8h pk;
#pragma unroll
                    for (int r = 0; r < 8; ++r) pk[r] = (h16)(lt[kt][r] * inv);
                    // lane owns column q; its 8 k's per tile are consecutive: one 16B store
                    *(v8h*)(sAttn + q * AT_STR + kt * 16 + hh * 8) = pk;
                }
            }
            // y^T[d][q] = sum_k gx[d][k] * attn[q][k] (f16 WMMA, M=4 pad 16, K=64)
            const int base0 = hh ? 16 : 0;
            const int grow = head * 4 + (l15 & 3);  // rows 4..15 duplicate (D rows discarded)
#pragma unroll 1
            for (int qt = 0; qt < 4; ++qt) {
                int q = qt * 16 + l15;
                v8f acc = zero8f();
#pragma unroll
                for (int k0 = 0; k0 < 64; k0 += 32) {
                    const h16* as = sGx + grow * 64 + k0 + base0;
                    v16h a = cat8(*(const v8h*)as, *(const v8h*)(as + 8));
                    const h16* bs = sAttn + q * AT_STR + k0 + hh * 16;
                    v16h b = cat8(*(const v8h*)bs, *(const v8h*)(bs + 8));
                    acc = wmma16(a, b, acc);
                }
                if (lane < 16) {              // valid rows d=0..3 in half 0
                    v4h yv;
#pragma unroll
                    for (int r = 0; r < 4; ++r) yv[r] = (h16)acc[r];
                    *(v4h*)(sYT + q * TP_STR + head * 4) = yv;  // 8B store, ic-contiguous
                }
            }
        }
    }
    __syncthreads();

    // ---- P5: out projection (M=512, N=64, K=256) + BN + ReLU + residual ----
    {
        float* on = out + (size_t)n * (C512 * P64);
        for (int job = w; job < 32 * 4; job += 8) {
            int mt = job >> 2, nt = job & 3;
            int m = mt * 16 + l15;
            int p = nt * 16 + l15;
            v8f acc = zero8f();
#pragma unroll 2
            for (int k0 = 0; k0 < IC256; k0 += 32) {
                const h16* as = outW + m * IC256 + k0 + (hh ? 16 : 0);
                v16h a = cat8(*(const v8h*)as, *(const v8h*)(as + 8));
                const h16* bs = sYT + p * TP_STR + k0 + hh * 16;
                v16h b = cat8(*(const v8h*)bs, *(const v8h*)(bs + 8));
                acc = wmma16(a, b, acc);
            }
#pragma unroll
            for (int r = 0; r < 8; ++r) {
                int mr = mt * 16 + r + hh * 8;
                float v = fmaxf(acc[r] * oSc[mr] + oSh[mr], 0.0f);
                on[mr * 64 + p] = v + qn[mr * 64 + p];   // fp32 residual from global
            }
        }
    }
}

extern "C" void kernel_launch(void* const* d_in, const int* in_sizes, int n_in,
                              void* d_out, int out_size, void* d_ws, size_t ws_size,
                              hipStream_t stream) {
    (void)in_sizes; (void)n_in; (void)out_size; (void)ws_size;
    const float* query   = (const float*)d_in[0];
    const float* context = (const float*)d_in[1];
    const float* pm_W    = (const float*)d_in[2];
    const float* pm_b    = (const float*)d_in[3];
    const float* g_W = (const float*)d_in[4];
    const float* g_g = (const float*)d_in[5],  *g_b = (const float*)d_in[6];
    const float* g_m = (const float*)d_in[7],  *g_v = (const float*)d_in[8];
    const float* t_W = (const float*)d_in[9];
    const float* t_g = (const float*)d_in[10], *t_b = (const float*)d_in[11];
    const float* t_m = (const float*)d_in[12], *t_v = (const float*)d_in[13];
    const float* p_W = (const float*)d_in[14];
    const float* p_g = (const float*)d_in[15], *p_b = (const float*)d_in[16];
    const float* p_m = (const float*)d_in[17], *p_v = (const float*)d_in[18];
    const float* o_W = (const float*)d_in[19];
    const float* o_g = (const float*)d_in[20], *o_b = (const float*)d_in[21];
    const float* o_m = (const float*)d_in[22], *o_v = (const float*)d_in[23];
    char* ws = (char*)d_ws;

    (void)hipFuncSetAttribute((const void*)lawin_main,
                              hipFuncAttributeMaxDynamicSharedMemorySize, SMEM_BYTES);

    lawin_prep<<<1024, 256, 0, stream>>>(pm_W, g_W, t_W, p_W, o_W,
                                         g_g, g_b, g_m, g_v,
                                         t_g, t_b, t_m, t_v,
                                         p_g, p_b, p_m, p_v,
                                         o_g, o_b, o_m, o_v, ws);
    lawin_main<<<NBATCH, 256, SMEM_BYTES, stream>>>(query, context, pm_b, ws,
                                                    (float*)d_out);
}
